// VisionTransformer_65549790872155
// MI455X (gfx1250) — matrix operand order repaired and checked by reading, not compiled, not run on hardware
//
#include <hip/hip_runtime.h>
#include <math.h>

// ---------------------------------------------------------------------------
// ViT forward for MI455X (gfx1250, wave32, WMMA).
// - all GEMMs via v_wmma_f32_16x16x32_bf16 (f32 accumulate)
// - weights pre-transposed/converted to bf16 [N,K] once -> B staging is a
//   contiguous 16B copy (no VALU in the GEMM inner loop)
// - interior tiles staged with global_load_async_to_lds_b128 (ASYNCcnt) and
//   double-buffered LDS so the DMA overlaps the WMMAs
// ---------------------------------------------------------------------------

typedef __attribute__((ext_vector_type(16))) __bf16 v16bf;
typedef __attribute__((ext_vector_type(8)))  __bf16 v8bf;
typedef __attribute__((ext_vector_type(8)))  float  v8f;

#define DEV_INLINE __device__ __forceinline__
#define USE_ASYNC_COPY 1

static constexpr int  B_   = 32;     // batch
static constexpr int  NT   = 197;    // tokens
static constexpr int  DM   = 768;    // model dim
static constexpr int  FFD  = 3072;   // ffn dim
static constexpr int  NP   = 224;    // tokens padded to 32 multiple
static constexpr int  NLAY = 12;
static constexpr long long RW = (long long)B_ * NT;   // 6304 rows

DEV_INLINE float gelu_exact(float x) {
    return 0.5f * x * (1.0f + erff(x * 0.70710678118654752f));
}

DEV_INLINE float gumbel_from_idx(unsigned long long idx) {
    unsigned long long z = idx * 0x9E3779B97F4A7C15ull + 0xD1B54A32D192ED03ull;
    z = (z ^ (z >> 30)) * 0xBF58476D1CE4E5B9ull;
    z = (z ^ (z >> 27)) * 0x94D049BB133111EBull;
    z ^= (z >> 31);
    float u = (float)((z >> 40) + 1ull) * (1.0f / 16777217.0f);
    return -__logf(-__logf(u));
}

DEV_INLINE void wait_async0() {
#if defined(__has_builtin)
#if __has_builtin(__builtin_amdgcn_s_wait_asynccnt)
    __builtin_amdgcn_s_wait_asynccnt(0);
    return;
#endif
#endif
#if USE_ASYNC_COPY
    asm volatile("s_wait_asynccnt 0" ::: "memory");
#endif
}

// 16B global -> LDS async copy (GV addressing, tracked with ASYNCcnt)
DEV_INLINE void async_cp16(const __bf16* gsrc, __bf16* ldst) {
#if USE_ASYNC_COPY
    unsigned lds = (unsigned)(unsigned long long)ldst;   // low 32b = LDS offset
    asm volatile("global_load_async_to_lds_b128 %0, %1, off"
                 :: "v"(lds), "v"(gsrc) : "memory");
#else
    *(uint4*)ldst = *(const uint4*)gsrc;
#endif
}

// A fragment: 16x32 bf16 (ISA table: lanes 0-15 K0..7,16..23; 16-31 K8..15,24..31)
DEV_INLINE v16bf load_frag_a(const __bf16* sA, int rowBase, int lane) {
    const __bf16* p = sA + (rowBase + (lane & 15)) * 32 + ((lane < 16) ? 0 : 8);
    v8bf lo = *(const v8bf*)p;
    v8bf hi = *(const v8bf*)(p + 16);
    v16bf r;
#pragma unroll
    for (int j = 0; j < 8; ++j) { r[j] = lo[j]; r[j + 8] = hi[j]; }
    return r;
}

// B fragment from [n][k] LDS tile (k stride 32)
DEV_INLINE v16bf load_frag_b(const __bf16* sB, int colBase, int lane) {
    const __bf16* p = sB + (colBase + (lane & 15)) * 32 + ((lane < 16) ? 0 : 8);
    v8bf lo = *(const v8bf*)p;
    v8bf hi = *(const v8bf*)(p + 16);
    v16bf r;
#pragma unroll
    for (int j = 0; j < 8; ++j) { r[j] = lo[j]; r[j + 8] = hi[j]; }
    return r;
}

DEV_INLINE void mma_step(const __bf16* pa, const __bf16* pb,
                         int wm, int wn, int lane, v8f (&acc)[2][2]) {
    v16bf af[2], bfr[2];
#pragma unroll
    for (int mi = 0; mi < 2; ++mi) af[mi] = load_frag_a(pa, wm * 32 + mi * 16, lane);
#pragma unroll
    for (int ni = 0; ni < 2; ++ni) bfr[ni] = load_frag_b(pb, wn * 32 + ni * 16, lane);
#pragma unroll
    for (int mi = 0; mi < 2; ++mi)
#pragma unroll
        for (int ni = 0; ni < 2; ++ni)
            acc[mi][ni] = __builtin_amdgcn_wmma_f32_16x16x32_bf16(
                false, af[mi], false, bfr[ni], (short)0, acc[mi][ni], false, false);
}

// ---------------------------------------------------------------------------
// Unified WMMA GEMM:  C[M,N] = A[M,K](bf16,row) @ Bt[N,K](bf16,row)^T
//   EPI : 0 = f32 out, 1 = bias + bf16 out, 2 = bias + GELU + bf16 out,
//         3 = bias + f32 out
//   CT  : 1 = write C transposed (bf16, Cb[c*ldc + r]) - used to produce V^T
// Block tile 128x64, 8 waves, wave tile 32x32 (2x2 WMMA).
// Interior blocks: async double-buffered staging; edge blocks: sync + bounds.
// ---------------------------------------------------------------------------
template <int EPI, int CT>
__global__ __launch_bounds__(256)
void gemm_wmma(const __bf16* __restrict__ A, const __bf16* __restrict__ Bt,
               const float* __restrict__ bias,
               float* __restrict__ Cf, __bf16* __restrict__ Cb,
               int M, int N, int K, int lda, int ldb, int ldc,
               long long strA, long long strB, long long strC) {
    __shared__ alignas(16) __bf16 sA[2][128 * 32];
    __shared__ alignas(16) __bf16 sB[2][64 * 32];

    const int t = threadIdx.x, lane = t & 31, wv = t >> 5;
    const int wm = wv >> 1, wn = wv & 1;
    const int m0 = blockIdx.y * 128, n0 = blockIdx.x * 64;
    const long long z = blockIdx.z;
    A  += z * strA;
    Bt += z * strB;

    // per-thread staging coordinates (16B granules)
    const int ar0 = t >> 2, akc0 = (t & 3) << 3;           // A chunk 0
    const int ar1 = (t + 256) >> 2, akc1 = ((t + 256) & 3) << 3;   // A chunk 1
    const int br = t >> 2, bkc = (t & 3) << 3;             // B chunk

    v8f acc[2][2] = {};
    const int nk = K >> 5;
    const bool full = (m0 + 128 <= M) && (n0 + 64 <= N);

    if (full) {
        // ---------------- async, double-buffered ----------------
        auto stage = [&](int k0, int buf) {
            async_cp16(A + (long long)(m0 + ar0) * lda + (k0 + akc0), &sA[buf][ar0 * 32 + akc0]);
            async_cp16(A + (long long)(m0 + ar1) * lda + (k0 + akc1), &sA[buf][ar1 * 32 + akc1]);
            async_cp16(Bt + (long long)(n0 + br) * ldb + (k0 + bkc), &sB[buf][br * 32 + bkc]);
        };
        stage(0, 0);
        for (int i = 0; i < nk; ++i) {
            wait_async0();          // my buf[i&1] granules are in LDS
            __syncthreads();        // everyone's are; previous compute done
            if (i + 1 < nk) stage((i + 1) << 5, (i + 1) & 1);
            mma_step(&sA[i & 1][0], &sB[i & 1][0], wm, wn, lane, acc);
        }
    } else {
        // ---------------- sync, bounds-checked ----------------
        for (int i = 0; i < nk; ++i) {
            const int k0 = i << 5;
            __syncthreads();
            uint4 v0 = make_uint4(0u, 0u, 0u, 0u), v1 = v0, v2 = v0;
            if (m0 + ar0 < M) v0 = *(const uint4*)(A + (long long)(m0 + ar0) * lda + (k0 + akc0));
            if (m0 + ar1 < M) v1 = *(const uint4*)(A + (long long)(m0 + ar1) * lda + (k0 + akc1));
            if (n0 + br < N)  v2 = *(const uint4*)(Bt + (long long)(n0 + br) * ldb + (k0 + bkc));
            *(uint4*)(&sA[0][ar0 * 32 + akc0]) = v0;
            *(uint4*)(&sA[0][ar1 * 32 + akc1]) = v1;
            *(uint4*)(&sB[0][br * 32 + bkc]) = v2;
            __syncthreads();
            if (i + 1 < nk)
                __builtin_prefetch(A + (long long)(m0 + wv * 16) * lda + (k0 + 32), 0, 1);
            mma_step(&sA[0][0], &sB[0][0], wm, wn, lane, acc);
        }
    }

    // ---------------- epilogue ----------------
    const int rb = m0 + wm * 32 + ((lane < 16) ? 0 : 8);
    const int cb = n0 + wn * 32 + (lane & 15);
    float*  CfP = Cf ? Cf + z * strC : nullptr;
    __bf16* CbP = Cb ? Cb + z * strC : nullptr;
#pragma unroll
    for (int mi = 0; mi < 2; ++mi) {
#pragma unroll
        for (int ni = 0; ni < 2; ++ni) {
            int c = cb + ni * 16;
            if (c >= N) continue;
            float bval = (EPI >= 1) ? bias[c] : 0.0f;
#pragma unroll
            for (int v = 0; v < 8; ++v) {
                int r = rb + mi * 16 + v;
                if (r >= M) continue;
                float xv = acc[mi][ni][v] + bval;
                if (EPI == 2) xv = gelu_exact(xv);
                if (CT)                      CbP[(long long)c * ldc + r] = (__bf16)xv;
                else if (EPI == 0 || EPI == 3) CfP[(long long)r * ldc + c] = xv;
                else                         CbP[(long long)r * ldc + c] = (__bf16)xv;
            }
        }
    }
}

// ---------------------------------------------------------------------------
// Patch embedding WMMA GEMM: rows = 32*196 patches, K = N = 768.
// Gather addresses hoisted out of the K loop; weights pre-transposed bf16.
// ---------------------------------------------------------------------------
__global__ __launch_bounds__(256)
void patch_embed_wmma(const float* __restrict__ X, const __bf16* __restrict__ PWt,
                      const float* __restrict__ PB, const float* __restrict__ POS,
                      float* __restrict__ Yf, __bf16* __restrict__ Yb) {
    __shared__ alignas(16) __bf16 sA[128 * 32];
    __shared__ alignas(16) __bf16 sB[64 * 32];

    const int t = threadIdx.x, lane = t & 31, wv = t >> 5;
    const int wm = wv >> 1, wn = wv & 1;
    const int m0 = blockIdx.y * 128, n0 = blockIdx.x * 64;   // 6272 rows exact

    // hoisted per-row image base offsets (k-independent)
    const int kk = t & 31;
    int rowbase[16];
#pragma unroll
    for (int i = 0; i < 16; ++i) {
        int rr = (t >> 5) + (i << 3);
        int r = m0 + rr;
        int b = r / 196, pr = r - b * 196;
        int hh = pr / 14, wp = pr - hh * 14;
        rowbase[i] = ((b * 3) * 224 + hh * 16) * 224 + wp * 16;
    }
    const int br = t >> 2, bkc = (t & 3) << 3;

    v8f acc[2][2] = {};
    for (int k0 = 0; k0 < 768; k0 += 32) {
        __syncthreads();
        {
            const int k = k0 + kk;
            const int c = k >> 8, rem = k & 255, pp = rem >> 4, qq = rem & 15;
            const int koff = (c * 224 + pp) * 224 + qq;
#pragma unroll
            for (int i = 0; i < 16; ++i)
                sA[((t >> 5) + (i << 3)) * 32 + kk] = (__bf16)X[rowbase[i] + koff];
            *(uint4*)(&sB[br * 32 + bkc]) =
                *(const uint4*)(PWt + (long long)(n0 + br) * 768 + (k0 + bkc));
        }
        __syncthreads();
        mma_step(&sA[0], &sB[0], wm, wn, lane, acc);
    }

    const int rb = m0 + wm * 32 + ((lane < 16) ? 0 : 8);
    const int cb = n0 + wn * 32 + (lane & 15);
#pragma unroll
    for (int mi = 0; mi < 2; ++mi) {
#pragma unroll
        for (int ni = 0; ni < 2; ++ni) {
            int cdx = cb + ni * 16;
            float bval = PB[cdx];
#pragma unroll
            for (int v = 0; v < 8; ++v) {
                int r = rb + mi * 16 + v;
                int b = r / 196, pr = r - b * 196;
                long long orow = (long long)(b * 197 + 1 + pr) * 768;
                float xv = acc[mi][ni][v] + bval + POS[(long long)(1 + pr) * 768 + cdx];
                Yf[orow + cdx] = xv;
                Yb[orow + cdx] = (__bf16)xv;
            }
        }
    }
}

// tiled transpose + f32->bf16 convert: src [K,N] f32 -> dst [N,K] bf16 (z batched)
__global__ __launch_bounds__(256)
void transpose_cvt(const float* __restrict__ src, __bf16* __restrict__ dst, int K, int N) {
    __shared__ float tile[32][33];
    const int k0 = blockIdx.x * 32, n0 = blockIdx.y * 32;
    const long long zoff = (long long)blockIdx.z * K * N;
    src += zoff; dst += zoff;
    const int tx = threadIdx.x & 31, ty = threadIdx.x >> 5;
#pragma unroll
    for (int i = 0; i < 32; i += 8)
        tile[ty + i][tx] = src[(long long)(k0 + ty + i) * N + (n0 + tx)];
    __syncthreads();
#pragma unroll
    for (int i = 0; i < 32; i += 8)
        dst[(long long)(n0 + ty + i) * K + (k0 + tx)] = (__bf16)tile[tx][ty + i];
}

// cls token row = cls + pos_embed[0]
__global__ void cls_pos_kernel(const float* __restrict__ CLS, const float* __restrict__ POS,
                               float* __restrict__ Yf, __bf16* __restrict__ Yb) {
    int b = blockIdx.x, t = threadIdx.x;
#pragma unroll
    for (int i = 0; i < 3; ++i) {
        int c = t + i * 256;
        float v = CLS[c] + POS[c];
        long long o = (long long)b * 197 * 768 + c;
        Yf[o] = v; Yb[o] = (__bf16)v;
    }
}

// fused residual + LayerNorm -> f32 (residual chain) + bf16 (next GEMM A)
__global__ __launch_bounds__(256)
void ln_kernel(const float* __restrict__ X, const float* __restrict__ R,
               const float* __restrict__ g, const float* __restrict__ bta,
               float* __restrict__ Yf, __bf16* __restrict__ Yb) {
    const int row = blockIdx.x, t = threadIdx.x;
    const float* xr = X + (long long)row * 768;
    const float* rr = R ? R + (long long)row * 768 : nullptr;
    float v[3], s1 = 0.f, s2 = 0.f;
#pragma unroll
    for (int i = 0; i < 3; ++i) {
        int c = t + i * 256;
        float x = xr[c] + (rr ? rr[c] : 0.f);
        v[i] = x; s1 += x; s2 += x * x;
    }
    __shared__ float r1[256], r2[256];
    r1[t] = s1; r2[t] = s2; __syncthreads();
    for (int s = 128; s > 0; s >>= 1) {
        if (t < s) { r1[t] += r1[t + s]; r2[t] += r2[t + s]; }
        __syncthreads();
    }
    float mean = r1[0] * (1.0f / 768.0f);
    float var  = r2[0] * (1.0f / 768.0f) - mean * mean;
    float inv  = rsqrtf(var + 1e-5f);
#pragma unroll
    for (int i = 0; i < 3; ++i) {
        int c = t + i * 256;
        float y = (v[i] - mean) * inv * g[c] + bta[c];
        long long o = (long long)row * 768 + c;
        Yf[o] = y;
        if (Yb) Yb[o] = (__bf16)y;
    }
}

// softmax with scale + Gumbel noise -> bf16 probs, zero-padded to 224 cols
__global__ __launch_bounds__(256)
void softmax_gumbel(const float* __restrict__ S, __bf16* __restrict__ P, int layer) {
    const int n = blockIdx.x, b = blockIdx.y, t = threadIdx.x;
    const float inv_sqrt_d = 0.036084391824351613f;
    const long long base = ((long long)b * 197 + n) * 224;
    float x = -3.0e38f;
    if (t < 197) {
        unsigned long long idx =
            ((((unsigned long long)layer * 32 + b) * 197 + n) * 197 + t);
        x = S[base + t] * inv_sqrt_d + gumbel_from_idx(idx);
    }
    __shared__ float red[256];
    red[t] = x; __syncthreads();
    for (int s = 128; s > 0; s >>= 1) { if (t < s) red[t] = fmaxf(red[t], red[t + s]); __syncthreads(); }
    float mx = red[0]; __syncthreads();
    float e = (t < 197) ? __expf(x - mx) : 0.f;
    red[t] = e; __syncthreads();
    for (int s = 128; s > 0; s >>= 1) { if (t < s) red[t] += red[t + s]; __syncthreads(); }
    float inv_sum = 1.0f / red[0];
    if (t < 224) P[base + t] = (__bf16)((t < 197) ? e * inv_sum : 0.0f);
}

__global__ void zero_bf16(__bf16* __restrict__ p, long long n) {
    long long i = (long long)blockIdx.x * blockDim.x + threadIdx.x;
    if (i < n) p[i] = (__bf16)0.0f;
}

// tiny classifier head: 32 x 100
__global__ void head_kernel(const float* __restrict__ Xf, const float* __restrict__ HW,
                            const float* __restrict__ HB, float* __restrict__ OUT) {
    int b = blockIdx.x, t = threadIdx.x;
    if (t < 100) {
        const float* xr = Xf + (long long)b * 197 * 768;
        float a = HB[t];
        for (int d = 0; d < 768; ++d) a += xr[d] * HW[d * 100 + t];
        OUT[b * 100 + t] = a;
    }
}

// ---------------------------------------------------------------------------
extern "C" void kernel_launch(void* const* d_in, const int* in_sizes, int n_in,
                              void* d_out, int out_size, void* d_ws, size_t ws_size,
                              hipStream_t stream) {
    (void)in_sizes; (void)n_in; (void)out_size; (void)ws_size;
    const float* x    = (const float*)d_in[0];
    const float* pw   = (const float*)d_in[1];
    const float* pb_  = (const float*)d_in[2];
    const float* cls  = (const float*)d_in[3];
    const float* pos  = (const float*)d_in[4];
    const float* Wq   = (const float*)d_in[5];
    const float* bq   = (const float*)d_in[6];
    const float* Wk   = (const float*)d_in[7];
    const float* bk   = (const float*)d_in[8];
    const float* Wv   = (const float*)d_in[9];
    const float* bv   = (const float*)d_in[10];
    const float* W1   = (const float*)d_in[11];
    const float* b1   = (const float*)d_in[12];
    const float* W2   = (const float*)d_in[13];
    const float* b2   = (const float*)d_in[14];
    const float* g1   = (const float*)d_in[15];
    const float* be1  = (const float*)d_in[16];
    const float* g2   = (const float*)d_in[17];
    const float* be2  = (const float*)d_in[18];
    const float* gf   = (const float*)d_in[19];
    const float* bff  = (const float*)d_in[20];
    const float* hw   = (const float*)d_in[21];
    const float* hbp  = (const float*)d_in[22];
    float* out = (float*)d_out;

    char* wsb = (char*)d_ws;
    size_t off = 0;
    auto take = [&](size_t bytes) -> void* {
        void* p = wsb + off;
        off = (off + bytes + 255) & ~(size_t)255;
        return p;
    };
    float*  x0f  = (float*) take((size_t)RW * DM * 4);
    __bf16* x0b  = (__bf16*)take((size_t)RW * DM * 2);
    float*  x1f  = (float*) take((size_t)RW * DM * 4);
    __bf16* x1b  = (__bf16*)take((size_t)RW * DM * 2);
    __bf16* qb_  = (__bf16*)take((size_t)RW * DM * 2);
    __bf16* kb_  = (__bf16*)take((size_t)RW * DM * 2);
    __bf16* vt_  = (__bf16*)take((size_t)B_ * DM * NP * 2);   // V^T, [768][224]/batch
    float*  sc   = (float*) take((size_t)B_ * NT * NP * 4);
    __bf16* pmat = (__bf16*)take((size_t)B_ * NT * NP * 2);
    float*  ao   = (float*) take((size_t)RW * DM * 4);
    __bf16* hbuf = (__bf16*)take((size_t)RW * FFD * 2);
    // pre-transposed bf16 weights [N,K]
    __bf16* wqT  = (__bf16*)take((size_t)NLAY * DM * DM * 2);
    __bf16* wkT  = (__bf16*)take((size_t)NLAY * DM * DM * 2);
    __bf16* wvT  = (__bf16*)take((size_t)NLAY * DM * DM * 2);
    __bf16* w1T  = (__bf16*)take((size_t)NLAY * DM * FFD * 2);
    __bf16* w2T  = (__bf16*)take((size_t)NLAY * DM * FFD * 2);
    __bf16* pwT  = (__bf16*)take((size_t)DM * DM * 2);

    // ---- one-time weight transpose/convert (HBM-bound, ~20us) ----
    transpose_cvt<<<dim3(DM / 32, DM / 32, NLAY), 256, 0, stream>>>(Wq, wqT, DM, DM);
    transpose_cvt<<<dim3(DM / 32, DM / 32, NLAY), 256, 0, stream>>>(Wk, wkT, DM, DM);
    transpose_cvt<<<dim3(DM / 32, DM / 32, NLAY), 256, 0, stream>>>(Wv, wvT, DM, DM);
    transpose_cvt<<<dim3(DM / 32, FFD / 32, NLAY), 256, 0, stream>>>(W1, w1T, DM, FFD);
    transpose_cvt<<<dim3(FFD / 32, DM / 32, NLAY), 256, 0, stream>>>(W2, w2T, FFD, DM);
    transpose_cvt<<<dim3(DM / 32, DM / 32, 1), 256, 0, stream>>>(pw, pwT, DM, DM);

    // zero V^T once (pad cols 197..223 stay zero; real cols rewritten each layer)
    {
        long long nv = (long long)B_ * DM * NP;
        zero_bf16<<<dim3((unsigned)((nv + 255) / 256)), 256, 0, stream>>>(vt_, nv);
    }

    // ---- patch embedding + cls/pos ----
    patch_embed_wmma<<<dim3(DM / 64, 6272 / 128, 1), 256, 0, stream>>>(
        x, pwT, pb_, pos, x0f, x0b);
    cls_pos_kernel<<<dim3(B_), 256, 0, stream>>>(cls, pos, x0f, x0b);

    const int mblk  = (int)((RW + 127) / 128);   // 50
    const int mblkB = (NT + 127) / 128;          // 2

    for (int l = 0; l < NLAY; ++l) {
        const __bf16* WqT_l = wqT + (size_t)l * DM * DM;
        const __bf16* WkT_l = wkT + (size_t)l * DM * DM;
        const __bf16* WvT_l = wvT + (size_t)l * DM * DM;
        const __bf16* W1T_l = w1T + (size_t)l * DM * FFD;
        const __bf16* W2T_l = w2T + (size_t)l * DM * FFD;
        const float* bq_l = bq + (size_t)l * DM;
        const float* bk_l = bk + (size_t)l * DM;
        const float* bv_l = bv + (size_t)l * DM;
        const float* b1_l = b1 + (size_t)l * FFD;
        const float* b2_l = b2 + (size_t)l * DM;

        // Q = x @ Wq + bq  (bf16)
        gemm_wmma<1, 0><<<dim3(DM / 64, mblk, 1), 256, 0, stream>>>(
            x0b, WqT_l, bq_l, nullptr, qb_, (int)RW, DM, DM, DM, DM, DM, 0, 0, 0);
        // K
        gemm_wmma<1, 0><<<dim3(DM / 64, mblk, 1), 256, 0, stream>>>(
            x0b, WkT_l, bk_l, nullptr, kb_, (int)RW, DM, DM, DM, DM, DM, 0, 0, 0);
        // V^T (batched; transposed-C epilogue writes Vt[d][token], ldc = 224)
        gemm_wmma<1, 1><<<dim3(DM / 64, mblkB, B_), 256, 0, stream>>>(
            x0b, WvT_l, bv_l, nullptr, vt_, NT, DM, DM, DM, DM, NP,
            (long long)NT * DM, 0, (long long)DM * NP);
        // scores = Q @ K^T (Bt = K rows), f32 out, ldc = 224
        gemm_wmma<0, 0><<<dim3((NT + 63) / 64, mblkB, B_), 256, 0, stream>>>(
            qb_, kb_, nullptr, sc, nullptr, NT, NT, DM, DM, DM, NP,
            (long long)NT * DM, (long long)NT * DM, (long long)NT * NP);
        // softmax(scale + gumbel) -> bf16 probs padded to 224
        softmax_gumbel<<<dim3(NT, B_), 256, 0, stream>>>(sc, pmat, l);
        // attn = P @ V  (Bt = V^T rows, K padded to 224)
        gemm_wmma<0, 0><<<dim3(DM / 64, mblkB, B_), 256, 0, stream>>>(
            pmat, vt_, nullptr, ao, nullptr, NT, DM, NP, NP, NP, DM,
            (long long)NT * NP, (long long)DM * NP, (long long)NT * DM);
        // LN1(x + attn)
        ln_kernel<<<dim3((unsigned)RW), 256, 0, stream>>>(
            x0f, ao, g1 + (size_t)l * DM, be1 + (size_t)l * DM, x1f, x1b);
        // FF1: gelu(x @ W1 + b1) -> bf16
        gemm_wmma<2, 0><<<dim3(FFD / 64, mblk, 1), 256, 0, stream>>>(
            x1b, W1T_l, b1_l, nullptr, hbuf, (int)RW, FFD, DM, DM, DM, FFD, 0, 0, 0);
        // FF2: h @ W2 + b2 -> f32
        gemm_wmma<3, 0><<<dim3(DM / 64, mblk, 1), 256, 0, stream>>>(
            hbuf, W2T_l, b2_l, ao, nullptr, (int)RW, DM, FFD, FFD, FFD, DM, 0, 0, 0);
        // LN2(x + ff)
        ln_kernel<<<dim3((unsigned)RW), 256, 0, stream>>>(
            x1f, ao, g2 + (size_t)l * DM, be2 + (size_t)l * DM, x0f, x0b);
    }

    // final LN + head
    ln_kernel<<<dim3((unsigned)RW), 256, 0, stream>>>(x0f, nullptr, gf, bff, x1f, x1b);
    head_kernel<<<dim3(B_), 128, 0, stream>>>(x1f, hw, hbp, out);
}
